// MaskMultiHeadAttention_53025666237153
// MI455X (gfx1250) — matrix operand order repaired and checked by reading, not compiled, not run on hardware
//
#include <hip/hip_runtime.h>
#include <cstddef>
#include <cstdint>

// ---------------- problem constants ----------------
#define B_   4
#define SQ_  1024
#define SK_  1024
#define D_   1024
#define H_   16
#define DH_  64           // head dim
#define INV_TMP 10.0f     // 1/0.1
#define EPS_NORM 1e-12f

typedef __attribute__((ext_vector_type(16))) __bf16 v16bf;
typedef __attribute__((ext_vector_type(8)))  float  v8f;

union Frag16 { v16bf v; uint4 u[2]; };

// ---- A fragment (16x32 bf16, M x K): lane L holds row m=m0+(L&15),
// K = kb..kb+15, kb = 16*(L>>4). Two 16B loads.
__device__ __forceinline__ v16bf load_a_frag(const __bf16* __restrict__ A, int lda,
                                             int m0, int k0, int lane) {
  int m  = m0 + (lane & 15);
  int kb = (lane >> 4) * 16;
  const __bf16* p = A + (size_t)m * lda + k0 + kb;
  Frag16 f;
  f.u[0] = *(const uint4*)(p);
  f.u[1] = *(const uint4*)(p + 8);
  return f.v;
}

// ---- B fragment (32x16 bf16, K x N) sourced from row-major [N x K] storage:
// lane L holds column n=n0+(L&15), K = ks..ks+15, ks = k0 + 16*(L>>4). One 32B load.
__device__ __forceinline__ v16bf load_b_frag(const __bf16* __restrict__ Wt, int ldb,
                                             int n0, int k0, int lane) {
  int n  = n0 + (lane & 15);
  int ks = k0 + ((lane >> 4) << 4);
  return *(const v16bf*)(Wt + (size_t)n * ldb + ks);
}

__device__ __forceinline__ v8f wmma_bf16(v16bf a, v16bf b, v8f c) {
  return __builtin_amdgcn_wmma_f32_16x16x32_bf16(false, a, false, b, (short)0, c,
                                                 false, false);
}

// =====================================================================
// 1) fp32 -> bf16 conversion (grid-strided)
// =====================================================================
__global__ void cvt_f32_bf16(const float* __restrict__ src, __bf16* __restrict__ dst,
                             size_t n) {
  size_t i = (size_t)blockIdx.x * blockDim.x + threadIdx.x;
  size_t stride = (size_t)gridDim.x * blockDim.x;
  for (; i < n; i += stride) dst[i] = (__bf16)src[i];
}

// =====================================================================
// 2) GEMM: C[M,N] = A[M,K](bf16) @ W[N,K]^T(bf16) + bias (fp32 out)
//    block = 128 (4 waves). Wave tile 16(M) x 64(N): 4 accumulators
//    share one A fragment; next k-step's fragments are loaded before
//    the current WMMAs issue (software pipeline -> loads overlap math).
// =====================================================================
__global__ void gemm_bf16_nt(const __bf16* __restrict__ A, const __bf16* __restrict__ W,
                             const float* __restrict__ bias, float* __restrict__ C,
                             int M, int N, int K) {
  int lane = threadIdx.x & 31;
  int wave = threadIdx.x >> 5;
  int m0 = blockIdx.x * 16;
  int n0 = blockIdx.y * 256 + wave * 64;

  v8f acc[4] = {{}, {}, {}, {}};
  v16bf a_cur = load_a_frag(A, K, m0, 0, lane);
  v16bf b_cur[4];
#pragma unroll
  for (int ct = 0; ct < 4; ++ct) b_cur[ct] = load_b_frag(W, K, n0 + ct * 16, 0, lane);

  int k0 = 0;
  for (; k0 + 32 < K; k0 += 32) {
    int kn = k0 + 32;
    if (kn + 32 < K) {
      __builtin_prefetch(A + (size_t)(m0 + (lane & 15)) * K + kn + 32, 0, 3);
      __builtin_prefetch(W + (size_t)(n0 + (lane & 15)) * K + kn + 32, 0, 3);
    }
    // issue next-step loads first so they overlap the WMMAs below
    v16bf a_nxt = load_a_frag(A, K, m0, kn, lane);
    v16bf b_nxt[4];
#pragma unroll
    for (int ct = 0; ct < 4; ++ct) b_nxt[ct] = load_b_frag(W, K, n0 + ct * 16, kn, lane);
#pragma unroll
    for (int ct = 0; ct < 4; ++ct) acc[ct] = wmma_bf16(a_cur, b_cur[ct], acc[ct]);
    a_cur = a_nxt;
#pragma unroll
    for (int ct = 0; ct < 4; ++ct) b_cur[ct] = b_nxt[ct];
  }
#pragma unroll
  for (int ct = 0; ct < 4; ++ct) acc[ct] = wmma_bf16(a_cur, b_cur[ct], acc[ct]);

  int hi = lane >> 4;
#pragma unroll
  for (int ct = 0; ct < 4; ++ct) {
    int n = n0 + ct * 16 + (lane & 15);
    float bv = bias[n];
#pragma unroll
    for (int r = 0; r < 8; ++r) {
      int m = m0 + r + 8 * hi;
      C[(size_t)m * N + n] = acc[ct][r] + bv;
    }
  }
}

// =====================================================================
// 3) per-head L2 normalize fp32 -> bf16. One wave per 64-element chunk.
// =====================================================================
__global__ void l2norm_to_bf16(const float* __restrict__ src, __bf16* __restrict__ dst,
                               size_t nchunks) {
  int lane = threadIdx.x & 31;
  size_t chunk = (size_t)blockIdx.x * (blockDim.x >> 5) + (threadIdx.x >> 5);
  if (chunk >= nchunks) return;
  const float* p = src + chunk * DH_;
  float x0 = p[lane * 2], x1 = p[lane * 2 + 1];
  float ss = x0 * x0 + x1 * x1;
#pragma unroll
  for (int m = 1; m <= 16; m <<= 1) ss += __shfl_xor(ss, m, 32);
  float inv = 1.0f / fmaxf(sqrtf(ss), EPS_NORM);
  dst[chunk * DH_ + lane * 2]     = (__bf16)(x0 * inv);
  dst[chunk * DH_ + lane * 2 + 1] = (__bf16)(x1 * inv);
}

// =====================================================================
// 4) tiled transpose: vp (fp32, [B,SK,D]) -> vpT (bf16, [(b*H+h)*64+c][SK]).
//    32x32 tiles via padded LDS: coalesced reads AND writes.
//    grid = (B*H, SK/32, 2); block = (32,8).
// =====================================================================
__global__ void vp_transpose(const float* __restrict__ vp, __bf16* __restrict__ vpT) {
  __shared__ float tile[32][33];
  int bh = blockIdx.x;
  int b = bh / H_, h = bh % H_;
  int n0 = blockIdx.y * 32;
  int c0 = blockIdx.z * 32;
  int tx = threadIdx.x, ty = threadIdx.y;
#pragma unroll
  for (int rr = 0; rr < 4; ++rr) {
    int r = ty * 4 + rr;                       // row within n-tile
    tile[r][tx] = vp[((size_t)b * SK_ + n0 + r) * D_ + h * DH_ + c0 + tx];
  }
  __syncthreads();
#pragma unroll
  for (int rr = 0; rr < 4; ++rr) {
    int r = ty * 4 + rr;                       // row within c-tile
    vpT[((size_t)bh * DH_ + c0 + r) * SK_ + n0 + tx] = (__bf16)tile[tx][r];
  }
}

// =====================================================================
// 5) Pass 1: per-key-column softmax stats over the query axis (m).
//    grid = (B*H, SK/64); block = 128 (4 waves, 16 columns each).
//    K-fragments are loop-invariant -> hoisted; inner loop is
//    2 A-loads + 2 WMMAs + online max/sum.
// =====================================================================
__global__ void attn_stats(const __bf16* __restrict__ qn, const __bf16* __restrict__ kn,
                           float* __restrict__ colmax, float* __restrict__ colsum) {
  int lane = threadIdx.x & 31;
  int wave = threadIdx.x >> 5;
  int bh = blockIdx.x;
  int b = bh / H_, h = bh % H_;
  int n0 = blockIdx.y * 64 + wave * 16;
  const __bf16* Q  = qn + (size_t)b * SQ_ * D_ + h * DH_;
  const __bf16* Kb = kn + (size_t)b * SK_ * D_ + h * DH_;

  v16bf bk0 = load_b_frag(Kb, D_, n0, 0, lane);    // hoisted: invariant over m
  v16bf bk1 = load_b_frag(Kb, D_, n0, 32, lane);

  float rm = -1e30f, rs = 0.0f;
  for (int mt = 0; mt < SQ_ / 16; ++mt) {
    v16bf a0 = load_a_frag(Q, D_, mt * 16, 0, lane);
    v16bf a1 = load_a_frag(Q, D_, mt * 16, 32, lane);
    v8f s = {};
    s = wmma_bf16(a0, bk0, s);
    s = wmma_bf16(a1, bk1, s);
#pragma unroll
    for (int r = 0; r < 8; ++r) {
      float v = s[r] * INV_TMP;
      if (v > rm) { rs = rs * __expf(rm - v) + 1.0f; rm = v; }
      else        { rs += __expf(v - rm); }
    }
  }
  // merge lane L with L^16 (same column, other half of the rows)
  float om = __shfl_xor(rm, 16, 32);
  float os = __shfl_xor(rs, 16, 32);
  float nm = fmaxf(rm, om);
  float ns = rs * __expf(rm - nm) + os * __expf(om - nm);
  if (lane < 16) {
    int n = n0 + lane;
    colmax[(size_t)bh * SK_ + n] = nm;
    colsum[(size_t)bh * SK_ + n] = ns;
  }
}

// =====================================================================
// 6) Pass 2: recompute scores, apply softmax(axis=m)+mask, accumulate
//    y = attn @ V via WMMA (attn staged through per-wave LDS for the
//    A-layout), write y directly as bf16, and compute the mask-weighted
//    score statistic per (b,h,m).
//    grid = (B*H, SQ/64); block = 128 (4 waves, one 16-row m-tile each).
// =====================================================================
__global__ void attn_apply(const __bf16* __restrict__ qn, const __bf16* __restrict__ kn,
                           const __bf16* __restrict__ vpT, const int* __restrict__ mask,
                           const float* __restrict__ colmax, const float* __restrict__ colsum,
                           __bf16* __restrict__ yb, float* __restrict__ wtmp) {
  __shared__ __attribute__((aligned(32))) __bf16 smem[4][16 * 32];
  int lane = threadIdx.x & 31;
  int wave = threadIdx.x >> 5;
  int bh = blockIdx.x;
  int b = bh / H_, h = bh % H_;
  int m0 = (blockIdx.y * 4 + wave) * 16;

  const __bf16* Q  = qn  + (size_t)b * SQ_ * D_ + h * DH_;
  const __bf16* Kb = kn  + (size_t)b * SK_ * D_ + h * DH_;
  const __bf16* Vt = vpT + (size_t)bh * DH_ * SK_;
  const float*  cm = colmax + (size_t)bh * SK_;
  const float*  cs = colsum + (size_t)bh * SK_;

  v16bf aq0 = load_a_frag(Q, D_, m0, 0, lane);
  v16bf aq1 = load_a_frag(Q, D_, m0, 32, lane);

  v8f yacc[4] = {{}, {}, {}, {}};
  float wnum[8], wden[8];
#pragma unroll
  for (int r = 0; r < 8; ++r) { wnum[r] = 0.0f; wden[r] = 0.0f; }

  int ncol = lane & 15, hi = lane >> 4;
  __bf16* ld = smem[wave];

  for (int nc = 0; nc < SK_ / 32; ++nc) {
    int nbase = nc * 32;
    // scores for keys [nbase, nbase+16) and [nbase+16, nbase+32)
    v8f s0 = {}, s1 = {};
    s0 = wmma_bf16(aq0, load_b_frag(Kb, D_, nbase, 0, lane), s0);
    s0 = wmma_bf16(aq1, load_b_frag(Kb, D_, nbase, 32, lane), s0);
    s1 = wmma_bf16(aq0, load_b_frag(Kb, D_, nbase + 16, 0, lane), s1);
    s1 = wmma_bf16(aq1, load_b_frag(Kb, D_, nbase + 16, 32, lane), s1);

    int nA = nbase + ncol, nB = nbase + 16 + ncol;
    float mA = cm[nA], sA = cs[nA];
    float mB = cm[nB], sB = cs[nB];
#pragma unroll
    for (int r = 0; r < 8; ++r) {
      int m = m0 + r + 8 * hi;
      float sc0 = s0[r] * INV_TMP;
      float sc1 = s1[r] * INV_TMP;
      int mk0 = mask[((size_t)b * SQ_ + m) * SK_ + nA] != 0;
      int mk1 = mask[((size_t)b * SQ_ + m) * SK_ + nB] != 0;
      float a0 = mk0 ? __expf(sc0 - mA) / sA : 0.0f;
      float a1 = mk1 ? __expf(sc1 - mB) / sB : 0.0f;
      ld[(r + 8 * hi) * 32 + ncol]      = (__bf16)a0;
      ld[(r + 8 * hi) * 32 + 16 + ncol] = (__bf16)a1;
      wnum[r] += sc0 * (float)mk0 + sc1 * (float)mk1;
      wden[r] += (float)(mk0 + mk1);
    }
    // re-read attn tile in A layout (16 x 32 over keys), y += attn @ V
    v16bf aA = load_a_frag(ld, 32, 0, 0, lane);
#pragma unroll
    for (int ct = 0; ct < 4; ++ct) {
      v16bf bv = load_b_frag(Vt, SK_, ct * 16, nbase, lane);
      yacc[ct] = wmma_bf16(aA, bv, yacc[ct]);
    }
  }

  // store y directly as bf16 (feeds the output-projection GEMM)
#pragma unroll
  for (int ct = 0; ct < 4; ++ct) {
    int c = h * DH_ + ct * 16 + ncol;
#pragma unroll
    for (int r = 0; r < 8; ++r) {
      int m = m0 + r + 8 * hi;
      yb[((size_t)b * SQ_ + m) * D_ + c] = (__bf16)yacc[ct][r];
    }
  }
  // reduce mask-weighted sums across the 16 lanes of each half (over n)
#pragma unroll
  for (int xm = 1; xm <= 8; xm <<= 1) {
#pragma unroll
    for (int r = 0; r < 8; ++r) {
      wnum[r] += __shfl_xor(wnum[r], xm, 32);
      wden[r] += __shfl_xor(wden[r], xm, 32);
    }
  }
  if (ncol == 0) {
#pragma unroll
    for (int r = 0; r < 8; ++r) {
      int m = m0 + r + 8 * hi;
      wtmp[(size_t)bh * SQ_ + m] = wnum[r] / (wden[r] + 1e-9f);
    }
  }
}

// =====================================================================
// 7) weight: mean over heads -> d_out tail
// =====================================================================
__global__ void weight_reduce(const float* __restrict__ wtmp, float* __restrict__ wout) {
  size_t i = (size_t)blockIdx.x * blockDim.x + threadIdx.x;
  if (i >= (size_t)B_ * SQ_) return;
  int b = (int)(i / SQ_), m = (int)(i % SQ_);
  float s = 0.0f;
  for (int h = 0; h < H_; ++h) s += wtmp[((size_t)b * H_ + h) * SQ_ + m];
  wout[i] = s * (1.0f / H_);
}

// =====================================================================
// host-side orchestration
// =====================================================================
extern "C" void kernel_launch(void* const* d_in, const int* in_sizes, int n_in,
                              void* d_out, int out_size, void* d_ws, size_t ws_size,
                              hipStream_t stream) {
  (void)in_sizes; (void)n_in; (void)out_size; (void)ws_size;
  const float* q    = (const float*)d_in[0];
  const float* k    = (const float*)d_in[1];
  const float* v    = (const float*)d_in[2];
  const int*   mask = (const int*)  d_in[3];
  const float* Wq   = (const float*)d_in[4];
  const float* bq   = (const float*)d_in[5];
  const float* Wk   = (const float*)d_in[6];
  const float* bk   = (const float*)d_in[7];
  const float* Wv   = (const float*)d_in[8];
  const float* bv   = (const float*)d_in[9];
  const float* Wo   = (const float*)d_in[10];
  const float* bo   = (const float*)d_in[11];

  const size_t MS = (size_t)B_ * SQ_;        // 4096 rows
  const size_t NA = MS * D_;                 // 4M activation elements
  const size_t NW = (size_t)D_ * D_;         // 1M weight elements

  // workspace carve-out
  char* p = (char*)d_ws;
  auto take = [&](size_t bytes) { char* r = p; p += (bytes + 255) & ~(size_t)255; return r; };
  __bf16* qb   = (__bf16*)take(NA * 2);
  __bf16* kb   = (__bf16*)take(NA * 2);
  __bf16* vb   = (__bf16*)take(NA * 2);
  __bf16* Wqb  = (__bf16*)take(NW * 2);
  __bf16* Wkb  = (__bf16*)take(NW * 2);
  __bf16* Wvb  = (__bf16*)take(NW * 2);
  __bf16* Wob  = (__bf16*)take(NW * 2);
  float*  qp   = (float*) take(NA * 4);
  float*  kp   = (float*) take(NA * 4);
  float*  vp   = (float*) take(NA * 4);
  __bf16* qn   = (__bf16*)take(NA * 2);
  __bf16* kn   = (__bf16*)take(NA * 2);
  __bf16* vpT  = (__bf16*)take(NA * 2);
  float*  cmax = (float*) take((size_t)B_ * H_ * SK_ * 4);
  float*  csum = (float*) take((size_t)B_ * H_ * SK_ * 4);
  __bf16* yb   = (__bf16*)take(NA * 2);
  float*  wtmp = (float*) take((size_t)B_ * H_ * SQ_ * 4);

  float* out_y = (float*)d_out;              // [B,SQ,D]
  float* out_w = out_y + NA;                 // [B,SQ]

  // 1) convert inputs & weights to bf16
  cvt_f32_bf16<<<2048, 256, 0, stream>>>(q,  qb,  NA);
  cvt_f32_bf16<<<2048, 256, 0, stream>>>(k,  kb,  NA);
  cvt_f32_bf16<<<2048, 256, 0, stream>>>(v,  vb,  NA);
  cvt_f32_bf16<<<1024, 256, 0, stream>>>(Wq, Wqb, NW);
  cvt_f32_bf16<<<1024, 256, 0, stream>>>(Wk, Wkb, NW);
  cvt_f32_bf16<<<1024, 256, 0, stream>>>(Wv, Wvb, NW);
  cvt_f32_bf16<<<1024, 256, 0, stream>>>(Wo, Wob, NW);

  // 2) QKV projections (WMMA GEMM, 16x64 wave tile, pipelined)
  dim3 gg(MS / 16, D_ / 256);
  gemm_bf16_nt<<<gg, 128, 0, stream>>>(qb, Wqb, bq, qp, (int)MS, D_, D_);
  gemm_bf16_nt<<<gg, 128, 0, stream>>>(kb, Wkb, bk, kp, (int)MS, D_, D_);
  gemm_bf16_nt<<<gg, 128, 0, stream>>>(vb, Wvb, bv, vp, (int)MS, D_, D_);

  // 3) per-head L2 norm -> bf16 operands; 4) tiled V transpose
  size_t nchunks = MS * H_;
  l2norm_to_bf16<<<(unsigned)((nchunks + 7) / 8), 256, 0, stream>>>(qp, qn, nchunks);
  l2norm_to_bf16<<<(unsigned)((nchunks + 7) / 8), 256, 0, stream>>>(kp, kn, nchunks);
  vp_transpose<<<dim3(B_ * H_, SK_ / 32, DH_ / 32), dim3(32, 8), 0, stream>>>(vp, vpT);

  // 5) column softmax stats; 6) apply + y(bf16) + weight stats
  attn_stats<<<dim3(B_ * H_, SK_ / 64), 128, 0, stream>>>(qn, kn, cmax, csum);
  attn_apply<<<dim3(B_ * H_, SQ_ / 64), 128, 0, stream>>>(qn, kn, vpT, mask,
                                                          cmax, csum, yb, wtmp);

  // 7) output projection and weight mean
  gemm_bf16_nt<<<gg, 128, 0, stream>>>(yb, Wob, bo, out_y, (int)MS, D_, D_);
  weight_reduce<<<(unsigned)((MS + 255) / 256), 256, 0, stream>>>(wtmp, out_w);
}
